// AbsTransformerQ_23287312679400
// MI455X (gfx1250) — compile-verified, matrix operand order checked by reading, not run on hardware
//
#include <hip/hip_runtime.h>

typedef __attribute__((ext_vector_type(16))) _Float16 v16h;
typedef __attribute__((ext_vector_type(8)))  float    v8f;
typedef __attribute__((ext_vector_type(2)))  float    v2f;
typedef __attribute__((ext_vector_type(4)))  unsigned int u32x4;
typedef __attribute__((ext_vector_type(8)))  int      i32x8;
typedef __attribute__((ext_vector_type(4)))  int      i32x4;
typedef _Float16 h16;

#define NB   128          // batch
#define NT   512          // time steps
#define NH   3            // heads
#define NTOK (NB*NT)

__device__ __forceinline__ v8f wmma32f16(v16h a, v16h b, v8f c) {
  return __builtin_amdgcn_wmma_f32_16x16x32_f16(false, a, false, b, (short)0, c,
                                                false, false);
}

#if __has_builtin(__builtin_amdgcn_wmma_f32_16x16x4_f32)
#define HAVE_WMMA_F32X4 1
#endif

#if __has_builtin(__builtin_amdgcn_tensor_load_to_lds) && \
    __has_builtin(__builtin_amdgcn_s_wait_tensorcnt)
#define HAVE_TDM 1
// 1-D contiguous TDM copy: global -> LDS, nbytes multiple of 8.
// D# per CDNA5 ISA ch.8: group0 = {count/flags, lds_addr, global_addr, type=2},
// group1 = {data_size=8B, tensor_dim0=tile_dim0=n64, dim1=1, stride0=n64}.
// clang-23 toolchain: 6-arg builtin (g0, g1, g2, g3, g4, cpol).
__device__ __forceinline__ void tdm_load_1d(unsigned lds_off, const void* gsrc,
                                            unsigned nbytes) {
  unsigned long long ga = (unsigned long long)(uintptr_t)gsrc;
  unsigned n64 = nbytes >> 3;
  u32x4 g0;
  g0[0] = 1u;                                             // count=1 (valid)
  g0[1] = lds_off;                                        // lds_addr [63:32]
  g0[2] = (unsigned)(ga & 0xFFFFFFFFu);                   // global_addr[31:0]
  g0[3] = (unsigned)((ga >> 32) & 0x01FFFFFFu) | (2u << 30); // ga[56:32]|type=2
  i32x8 g1;
  g1[0] = (int)(3u << 16);                 // wg_mask=0, data_size=3 (8 bytes)
  g1[1] = (int)((n64 & 0xFFFFu) << 16);    // tensor_dim0[15:0] @ bits 63:48
  g1[2] = (int)((n64 >> 16) | (1u << 16)); // tensor_dim0[31:16], tensor_dim1=1
  g1[3] = (int)((n64 & 0xFFFFu) << 16);    // tile_dim0 @ bits 127:112
  g1[4] = 1;                               // tile_dim1=1, tile_dim2=0
  g1[5] = (int)n64;                        // tensor_dim0_stride[31:0]
  g1[6] = 0;
  g1[7] = 0;
  i32x4 z4 = {0, 0, 0, 0};
  i32x8 z8 = {0, 0, 0, 0, 0, 0, 0, 0};
  __builtin_amdgcn_tensor_load_to_lds(g0, g1, z4, z4, z8, 0);
}
#endif

// ---------------- kernel 0: weights -> f16 (W1 zero-padded 28->32) -----------
__global__ void k_prep_weights(const float* __restrict__ W1,
                               const float* __restrict__ W2,
                               const float* __restrict__ Wout,
                               h16* __restrict__ W1p, h16* __restrict__ W2h,
                               h16* __restrict__ Wouth) {
  const int total = 256*32 + 256*256 + 256;
  for (int idx = blockIdx.x*blockDim.x + threadIdx.x; idx < total;
       idx += gridDim.x*blockDim.x) {
    if (idx < 256*32) {
      int r = idx >> 5, c = idx & 31;
      W1p[idx] = (c < 28) ? (h16)W1[r*28 + c] : (h16)0.f;
    } else if (idx < 256*32 + 256*256) {
      int i = idx - 256*32;
      W2h[i] = (h16)W2[i];
    } else {
      int i = idx - (256*32 + 256*256);
      Wouth[i] = (h16)Wout[i];
    }
  }
}

// ---------------- kernel 1: per-token QKV + MLP-input staging ----------------
__global__ void __launch_bounds__(256)
k_token_prep(const float* __restrict__ state, const float* __restrict__ action,
             const float* __restrict__ Wq, const float* __restrict__ Wk,
             const float* __restrict__ Wv,
             float* __restrict__ Qf, float* __restrict__ Kf,
             h16* __restrict__ Vt, h16* __restrict__ Xh) {
  int tok = blockIdx.x*blockDim.x + threadIdx.x;
  if (tok >= NTOK) return;
  float qs[16];
  #pragma unroll
  for (int i = 0; i < 14; ++i) qs[i] = state[tok*14 + i];
  qs[14] = action[tok*2 + 0];
  qs[15] = action[tok*2 + 1];
  const int b  = tok >> 9;
  const int tt = tok & 511;
  #pragma unroll
  for (int i = 0; i < 12; ++i) {
    float aq = 0.f;
    #pragma unroll
    for (int j = 0; j < 16; ++j) aq += Wq[i*16 + j] * qs[j];
    Qf[tok*12 + i] = aq * 0.5f;                 // fold 1/sqrt(K) into Q
    float ak = 0.f, av = 0.f;
    #pragma unroll
    for (int j = 0; j < 4; ++j) {               // t_state = state[:, :, :4]
      ak += Wk[i*4 + j] * qs[j];
      av += Wv[i*4 + j] * qs[j];
    }
    Kf[tok*12 + i] = ak;
    int h = i >> 2, kk = i & 3;                 // V transposed: [b][h][kk][T]
    Vt[(((size_t)(b*NH + h))*4 + kk)*NT + tt] = (h16)av;
  }
  #pragma unroll
  for (int j = 0; j < 16; ++j) Xh[tok*32 + 12 + j] = (h16)qs[j];
  #pragma unroll
  for (int j = 0; j < 4; ++j)  Xh[tok*32 + 28 + j] = (h16)0.f;
}

// ---------------- kernel 2: attention, one wave per (b,h,16-query tile) -----
__global__ void __launch_bounds__(32)
k_attention(const float* __restrict__ Qf, const float* __restrict__ Kf,
            const h16* __restrict__ Vt, h16* __restrict__ Xh) {
  __shared__ float sc[16][520];   // 16 query rows x 512 keys (+pad)
  __shared__ float red[2][16];
  __shared__ float rs[16];

  const int qt = blockIdx.x, h = blockIdx.y, b = blockIdx.z;
  const int lane = threadIdx.x;
  const int m = lane & 15, hlf = lane >> 4;
  const int qbase = qt * 16;

#if defined(HAVE_WMMA_F32X4)
  // ---- S = (Q*scale) @ K^T via native 16x16x4 f32 WMMA (exact fit, fp32) ---
  // A 16x4 f32 layout: lane (m,half): vgpr0=K(2*half), vgpr1=K(2*half+1)
  v2f aQ;
  {
    const float* qp = Qf + ((size_t)(b*NT + qbase + m))*12 + h*4 + 2*hlf;
    aQ[0] = qp[0]; aQ[1] = qp[1];
  }
  for (int kb = 0; kb < 32; ++kb) {
    const float* kp = Kf + ((size_t)(b*NT + kb*16 + m))*12 + h*4 + 2*hlf;
    v2f bK; bK[0] = kp[0]; bK[1] = kp[1];
    v8f c = {};
    c = __builtin_amdgcn_wmma_f32_16x16x4_f32(false, aQ, false, bK, (short)0, c,
                                              false, false);
    #pragma unroll
    for (int j = 0; j < 8; ++j) sc[j + 8*hlf][kb*16 + m] = c[j];
  }
#else
  // ---- fallback: f16 WMMA with K padded 4->32 ----
  v16h aq;
  #pragma unroll
  for (int e = 0; e < 16; ++e) aq[e] = (h16)0.f;
  if (hlf == 0) {
    const float* qp = Qf + ((size_t)(b*NT + qbase + m))*12 + h*4;
    #pragma unroll
    for (int e = 0; e < 4; ++e) aq[e] = (h16)qp[e];
  }
  for (int kb = 0; kb < 32; ++kb) {
    v16h bk;
    #pragma unroll
    for (int e = 0; e < 16; ++e) bk[e] = (h16)0.f;
    if (hlf == 0) {
      const float* kp = Kf + ((size_t)(b*NT + kb*16 + m))*12 + h*4;
      #pragma unroll
      for (int e = 0; e < 4; ++e) bk[e] = (h16)kp[e];
    }
    v8f c = {};
    c = wmma32f16(aq, bk, c);
    #pragma unroll
    for (int j = 0; j < 8; ++j) sc[j + 8*hlf][kb*16 + m] = c[j];
  }
#endif
  __syncthreads();

  // ---- softmax: lane owns (row=m, column segment=hlf) ----
  {
    const int base = hlf * 256;
    float mx = -3.0e38f;
    for (int c2 = 0; c2 < 256; ++c2) mx = fmaxf(mx, sc[m][base + c2]);
    red[hlf][m] = mx;
    __syncthreads();
    float M = fmaxf(red[0][m], red[1][m]);
    __syncthreads();
    float s = 0.f;
    for (int c2 = 0; c2 < 256; ++c2) {
      float e = __expf(sc[m][base + c2] - M);
      sc[m][base + c2] = e;
      s += e;
    }
    red[hlf][m] = s;
    __syncthreads();
    if (hlf == 0) rs[m] = red[0][m] + red[1][m];
    __syncthreads();
  }

  // ---- O(16x4) = P(16x512) @ V(512x4): 16 accumulated f16 WMMAs ----
  v8f o = {};
  const bool vlive = (m < 4);
  const h16* vb = Vt + (((size_t)(b*NH + h))*4 + (m & 3))*NT;  // valid all lanes
  for (int ch = 0; ch < 16; ++ch) {
    const int k0 = ch * 32;
    v16h ap;
    #pragma unroll
    for (int e = 0; e < 8; ++e) ap[e]     = (h16)sc[m][k0 + 8*hlf + e];
    #pragma unroll
    for (int e = 0; e < 8; ++e) ap[8 + e] = (h16)sc[m][k0 + 16 + 8*hlf + e];
    v16h bv;                                  // branchless zero-select
    #pragma unroll
    for (int e = 0; e < 8; ++e) bv[e]     = vlive ? vb[k0 + 8*hlf + e] : (h16)0.f;
    #pragma unroll
    for (int e = 0; e < 8; ++e) bv[8 + e] = vlive ? vb[k0 + 16 + 8*hlf + e] : (h16)0.f;
    o = wmma32f16(ap, bv, o);                 // uniform, EXEC all-1s
  }

  if (vlive) {
    #pragma unroll
    for (int j = 0; j < 8; ++j) {
      const int r = j + 8*hlf;
      float val = o[j] / rs[r];
      Xh[((size_t)(b*NT + qbase + r))*32 + h*4 + m] = (h16)val;
    }
  }
}

// ---------------- kernel 3: fused MLP 32->256->256->1, 64-token tiles -------
__global__ void __launch_bounds__(256)
k_mlp(const h16* __restrict__ Xh, const h16* __restrict__ W1p,
      const float* __restrict__ bias1, const h16* __restrict__ W2h,
      const float* __restrict__ bias2, const h16* __restrict__ Wouth,
      const float* __restrict__ boutp, float* __restrict__ out) {
  __shared__ __align__(16) h16 Xt[64*32];     // 4 KB  (TDM-staged X tile)
  __shared__ __align__(16) h16 W1s[256*32];   // 16 KB (TDM-staged W1)
  __shared__ h16 H1[64][264];                 // 33.8 KB
  __shared__ float oacc[64];

  const int tid = threadIdx.x;
  const int wave = tid >> 5, lane = tid & 31;
  const int m16 = lane & 15, hlf = lane >> 4;
  const int mtile = wave >> 1;     // 4 M-tiles x 2 waves each
  const int nhalf = wave & 1;      // each wave owns 8 of 16 N-tiles
  const int tilebase = blockIdx.x * 64;
  const int mrow = mtile*16 + m16;

#if defined(HAVE_TDM)
  if (tid < 32) {                  // wave 0 drives the Tensor Data Mover
    tdm_load_1d((unsigned)(uintptr_t)&Xt[0],
                Xh + (size_t)tilebase*32, 64*32*2);
    tdm_load_1d((unsigned)(uintptr_t)&W1s[0], W1p, 256*32*2);
    __builtin_amdgcn_s_wait_tensorcnt(0);
  }
#else
  for (int i = tid; i < 64*32; i += 256) Xt[i] = Xh[(size_t)tilebase*32 + i];
  for (int i = tid; i < 256*32; i += 256) W1s[i] = W1p[i];
#endif
  if (tid < 64) oacc[tid] = 0.f;
  __syncthreads();

  // A1: 16x32 f16 rows of the MLP input, from LDS
  const h16* xp = &Xt[mrow*32];
  v16h a1;
  #pragma unroll
  for (int e = 0; e < 8; ++e) a1[e]     = xp[8*hlf + e];
  #pragma unroll
  for (int e = 0; e < 8; ++e) a1[8 + e] = xp[16 + 8*hlf + e];

  // layer 1: H1 = relu(X @ W1^T + b1)
  for (int nt = 0; nt < 8; ++nt) {
    const int n = (nhalf*8 + nt)*16 + m16;
    const h16* wp = &W1s[n*32];
    v16h bmat;
    #pragma unroll
    for (int e = 0; e < 8; ++e) bmat[e]     = wp[8*hlf + e];
    #pragma unroll
    for (int e = 0; e < 8; ++e) bmat[8 + e] = wp[16 + 8*hlf + e];
    v8f c = {};
    c = wmma32f16(a1, bmat, c);
    const float bb = bias1[n];
    #pragma unroll
    for (int j = 0; j < 8; ++j) {
      float v = c[j] + bb;
      v = v > 0.f ? v : 0.f;
      H1[mtile*16 + j + 8*hlf][n] = (h16)v;
    }
  }
  __syncthreads();

  // preload layer-2 A chunks from LDS once, reuse across all N tiles
  v16h a2[8];
  #pragma unroll
  for (int ch = 0; ch < 8; ++ch) {
    #pragma unroll
    for (int e = 0; e < 8; ++e) a2[ch][e]     = H1[mrow][ch*32 + 8*hlf + e];
    #pragma unroll
    for (int e = 0; e < 8; ++e) a2[ch][8 + e] = H1[mrow][ch*32 + 16 + 8*hlf + e];
  }

  float pacc[8];
  #pragma unroll
  for (int j = 0; j < 8; ++j) pacc[j] = 0.f;

  // layer 2 (+ fused 256->1 output projection in the epilogue)
  for (int nt = 0; nt < 8; ++nt) {
    const int n = (nhalf*8 + nt)*16 + m16;
    const h16* wp = W2h + (size_t)n*256;
    if (nt + 1 < 8)   // prefetch next W2 row block (global_prefetch_b8)
      __builtin_prefetch(W2h + (size_t)((nhalf*8 + nt + 1)*16 + m16)*256, 0, 1);
    v8f c = {};
    #pragma unroll
    for (int ch = 0; ch < 8; ++ch) {
      v16h bmat;
      #pragma unroll
      for (int e = 0; e < 8; ++e) bmat[e]     = wp[ch*32 + 8*hlf + e];
      #pragma unroll
      for (int e = 0; e < 8; ++e) bmat[8 + e] = wp[ch*32 + 16 + 8*hlf + e];
      c = wmma32f16(a2[ch], bmat, c);
    }
    const float bb = bias2[n];
    const float wo = (float)Wouth[n];
    #pragma unroll
    for (int j = 0; j < 8; ++j) {
      float v = c[j] + bb;
      v = v > 0.f ? v : 0.f;
      pacc[j] += v * wo;
    }
  }

  #pragma unroll
  for (int j = 0; j < 8; ++j)
    atomicAdd(&oacc[mtile*16 + j + 8*hlf], pacc[j]);   // ds_add_f32
  __syncthreads();

  if (tid < 64) out[tilebase + tid] = oacc[tid] + boutp[0];
}

extern "C" void kernel_launch(void* const* d_in, const int* in_sizes, int n_in,
                              void* d_out, int out_size, void* d_ws, size_t ws_size,
                              hipStream_t stream) {
  (void)in_sizes; (void)n_in; (void)out_size; (void)ws_size;
  const float* state  = (const float*)d_in[0];
  const float* action = (const float*)d_in[1];
  const float* Wq     = (const float*)d_in[2];
  const float* Wk     = (const float*)d_in[3];
  const float* Wv     = (const float*)d_in[4];
  const float* W1     = (const float*)d_in[5];
  const float* b1     = (const float*)d_in[6];
  const float* W2     = (const float*)d_in[7];
  const float* b2     = (const float*)d_in[8];
  const float* Wout   = (const float*)d_in[9];
  const float* bout   = (const float*)d_in[10];
  float* out = (float*)d_out;

  const size_t ntok = (size_t)NTOK;
  float* Qf   = (float*)d_ws;          // ntok*12 f32 (scale folded)
  float* Kf   = Qf + ntok*12;          // ntok*12 f32
  h16* Vt     = (h16*)(Kf + ntok*12);  // ntok*12 f16  ([b][h][k][T])
  h16* Xh     = Vt + ntok*12;          // ntok*32 f16  (padded MLP input)
  h16* W1p    = Xh + ntok*32;          // 256*32
  h16* W2h    = W1p + 256*32;          // 256*256
  h16* Wouth  = W2h + 256*256;         // 256
  // total workspace ~11.8 MB

  k_prep_weights<<<64, 256, 0, stream>>>(W1, W2, Wout, W1p, W2h, Wouth);
  k_token_prep<<<NTOK/256, 256, 0, stream>>>(state, action, Wq, Wk, Wv,
                                             Qf, Kf, Vt, Xh);
  k_attention<<<dim3(NT/16, NH, NB), 32, 0, stream>>>(Qf, Kf, Vt, Xh);
  k_mlp<<<NTOK/64, 256, 0, stream>>>(Xh, W1p, b1, W2h, b2, Wouth, bout, out);
}
